// AttentionLayer_76020921139225
// MI455X (gfx1250) — matrix operand order repaired and checked by reading, not compiled, not run on hardware
//
#include <hip/hip_runtime.h>

typedef __bf16 bf16;
typedef __attribute__((ext_vector_type(16))) __bf16 v16bf;
typedef __attribute__((ext_vector_type(8)))  __bf16 v8bf;
typedef __attribute__((ext_vector_type(8)))  float  v8f;

#define B_ 8
#define N_ 1024
#define D_ 256
#define H_ 4
#define M_ (B_ * N_)          // 8192 rows
#define EPS_ 1e-6f
#define LDT 40                // padded LDS row stride (bf16 elems), 80B rows keep 16B alignment

__device__ __forceinline__ bf16 f2bf(float f) { return (bf16)f; }
__device__ __forceinline__ float bf2f(bf16 b) { return (float)b; }

__device__ __forceinline__ v8f wmma_bf16(v16bf a, v16bf b, v8f c) {
  // D = A(16x32) * B(32x16) + C, fp32 accum
  return __builtin_amdgcn_wmma_f32_16x16x32_bf16(false, a, false, b, (short)0, c,
                                                 false, false);
}

// ---- CDNA5 async global->LDS copy (ASYNCcnt path), 16 bytes per lane ----
// GV mode: the global and LDS addresses both come from VGPRs plus the immediate offset,
// i.e. the immediate offset advances BOTH addresses — so a group of copies that
// walks LDS and global in lockstep shares one dst VGPR + one addr VGPR pair.
// Generic pointers to __shared__ carry the LDS byte offset in their low 32 bits.
__device__ __forceinline__ void async_ld_b128(bf16* lds, const bf16* g, int byte_ofs) {
  unsigned dst = (unsigned)(uintptr_t)lds;
  asm volatile("global_load_async_to_lds_b128 %0, %1, off offset:%2"
               :
               : "v"(dst), "v"(g), "i"(byte_ofs)
               : "memory");
}

__device__ __forceinline__ void wait_async0() {
  asm volatile("s_wait_asynccnt 0" ::: "memory");
}

// A fragment (16x32 bf16): lane L -> row m0 + L%16; elems 0..7 = K[k0+half*8..),
// elems 8..15 = K[k0+16+half*8..). Two 16-byte loads.
__device__ __forceinline__ v16bf load_frag_a(const bf16* base, int ld, int m0, int k0) {
  const int lane = threadIdx.x & 31;
  const int m = m0 + (lane & 15);
  const int hf = lane >> 4;
  const bf16* p = base + (size_t)m * ld + k0 + hf * 8;
  v8bf lo = *(const v8bf*)p;
  v8bf hi = *(const v8bf*)(p + 16);
  return __builtin_shufflevector(lo, hi, 0, 1, 2, 3, 4, 5, 6, 7, 8, 9, 10, 11, 12, 13, 14, 15);
}

// B fragment (32x16 bf16), source stored K-contiguous per column ("Bt"[n][k]):
// lane L -> col n0 + L%16; elems 0..7 = K[k0+half*8..), elems 8..15 = K[k0+16+half*8..). Two 16-byte loads.
__device__ __forceinline__ v16bf load_frag_b(const bf16* base, int ld, int n0, int k0) {
  const int lane = threadIdx.x & 31;
  const int n = n0 + (lane & 15);
  const int hf = lane >> 4;
  const bf16* p = base + (size_t)n * ld + k0 + hf * 8;
  v8bf lo = *(const v8bf*)p;
  v8bf hi = *(const v8bf*)(p + 16);
  return __builtin_shufflevector(lo, hi, 0, 1, 2, 3, 4, 5, 6, 7, 8, 9, 10, 11, 12, 13, 14, 15);
}

// -------- shared GEMM mainloop: C[128x128] per WG, 8 waves, wave = 32x64 ----------
// Double-buffered LDS tiles fed by async global->LDS copies: while WMMAs consume
// buffer p, tile k+1 streams into buffer 1-p on the ASYNCcnt path.
// As/Bs must each hold 2*128*LDT bf16.
__device__ __forceinline__ void gemm_mainloop(const bf16* __restrict__ A, int lda,
                                              const bf16* __restrict__ Bt, int ldb,
                                              int K, int m_blk, int n_blk,
                                              bf16* As, bf16* Bs, v8f acc[8]) {
  const int tid = threadIdx.x;
  const int wid = tid >> 5;
  const int wm = wid >> 1;        // 0..3  (32-row strips)
  const int wn = wid & 1;         // 0..1  (64-col strips)
  const int row = tid >> 1;       // 0..127
  const int col = (tid & 1) << 4; // 0 or 16
  const int T = K >> 5;           // number of 32-wide K tiles

  const bf16* gaRow = A + (size_t)(m_blk + row) * lda + col;
  const bf16* gbRow = Bt + (size_t)(n_blk + row) * ldb + col;
  bf16* sa = As + row * LDT + col;
  bf16* sb = Bs + row * LDT + col;

  // prime buffer 0 with tile 0
  async_ld_b128(sa, gaRow, 0);
  async_ld_b128(sa, gaRow, 16);
  async_ld_b128(sb, gbRow, 0);
  async_ld_b128(sb, gbRow, 16);

  for (int i = 0; i < T; ++i) {
    wait_async0();     // this wave's writes into buffer i&1 are done
    __syncthreads();   // ...and every other wave's too; prior compute also done
    if (i + 1 < T) {
      const int kn = (i + 1) << 5;
      const int pb = ((i + 1) & 1) * (128 * LDT);
      const bf16* gaK = gaRow + kn;
      const bf16* gbK = gbRow + kn;
      bf16* saP = sa + pb;
      bf16* sbP = sb + pb;
      async_ld_b128(saP, gaK, 0);
      async_ld_b128(saP, gaK, 16);
      async_ld_b128(sbP, gbK, 0);
      async_ld_b128(sbP, gbK, 16);
    }
    const int pc = (i & 1) * (128 * LDT);
    v16bf afr0 = load_frag_a(As + pc, LDT, wm * 32, 0);
    v16bf afr1 = load_frag_a(As + pc, LDT, wm * 32 + 16, 0);
#pragma unroll
    for (int j = 0; j < 4; ++j) {
      v16bf bfr = load_frag_b(Bs + pc, LDT, wn * 64 + j * 16, 0);
      acc[j]     = wmma_bf16(afr0, bfr, acc[j]);
      acc[4 + j] = wmma_bf16(afr1, bfr, acc[4 + j]);
    }
  }
}

// -------- weight prep: fp32 (H,D,D)[h][d][e] -> bf16 transposed [h][e][d] --------
__global__ __launch_bounds__(256) void transpose_w_kernel(const float* __restrict__ w,
                                                          bf16* __restrict__ wt) {
  int idx = blockIdx.x * 256 + threadIdx.x;   // < H*D*D = 262144
  int h = idx >> 16;
  int r = idx & 65535;
  int d = r >> 8;
  int e = r & 255;
  wt[(h << 16) + (e << 8) + d] = f2bf(w[idx]);
}

// out_w (D*H=1024, D=256) -> bf16 transposed (256, 1024)
__global__ __launch_bounds__(256) void transpose_ow_kernel(const float* __restrict__ w,
                                                           bf16* __restrict__ wt) {
  int idx = blockIdx.x * 256 + threadIdx.x;   // < 1024*256
  int kk = idx >> 8;
  int n = idx & 255;
  wt[n * 1024 + kk] = f2bf(w[idx]);
}

// -------- block reductions for LayerNorm (256 threads == D) --------
__device__ __forceinline__ float block_reduce_sum(float v, float* red) {
  int t = threadIdx.x;
  red[t] = v;
  __syncthreads();
  for (int s = 128; s > 0; s >>= 1) {
    if (t < s) red[t] += red[t + s];
    __syncthreads();
  }
  float r = red[0];
  __syncthreads();
  return r;
}

__global__ __launch_bounds__(256) void ln_x_kernel(const float* __restrict__ x,
                                                   const float* __restrict__ g,
                                                   const float* __restrict__ bb,
                                                   bf16* __restrict__ xn) {
  __shared__ float red[256];
  int row = blockIdx.x, t = threadIdx.x;
  float v = x[(size_t)row * D_ + t];
  float mu = block_reduce_sum(v, red) * (1.0f / D_);
  float d = v - mu;
  float var = block_reduce_sum(d * d, red) * (1.0f / D_);
  xn[(size_t)row * D_ + t] = f2bf(d * rsqrtf(var + EPS_) * g[t] + bb[t]);
}

// -------- 16 fused projections: z = h*4 + {q,k,v,gate} --------
__global__ __launch_bounds__(256) void proj_gemm_kernel(
    const bf16* __restrict__ xn, const bf16* __restrict__ wqT,
    const bf16* __restrict__ wkT, const bf16* __restrict__ wvT,
    const bf16* __restrict__ wgT, bf16* __restrict__ qo, bf16* __restrict__ ko,
    bf16* __restrict__ vTo, bf16* __restrict__ gateo) {
  __shared__ __align__(16) bf16 As[2 * 128 * LDT];
  __shared__ __align__(16) bf16 Bs[2 * 128 * LDT];
  const int z = blockIdx.z;
  const int h = z >> 2, which = z & 3;
  const int m_blk = blockIdx.y * 128;
  const int n_blk = blockIdx.x * 128;
  const bf16* Bt = (which == 0) ? wqT : (which == 1) ? wkT : (which == 2) ? wvT : wgT;
  Bt += (size_t)h * D_ * D_;

  v8f acc[8];
  v8f zero = {};
#pragma unroll
  for (int i = 0; i < 8; ++i) acc[i] = zero;
  gemm_mainloop(xn, D_, Bt, D_, D_, m_blk, n_blk, As, Bs, acc);

  const int tid = threadIdx.x;
  const int wid = tid >> 5, wm = wid >> 1, wn = wid & 1;
  const int lane = tid & 31, nloc = lane & 15, hf = lane >> 4;
#pragma unroll
  for (int i = 0; i < 2; ++i) {
#pragma unroll
    for (int j = 0; j < 4; ++j) {
      v8f a = acc[i * 4 + j];
      int gm0 = m_blk + wm * 32 + i * 16 + hf * 8;
      int gn = n_blk + wn * 64 + j * 16 + nloc;
#pragma unroll
      for (int r = 0; r < 8; ++r) {
        int gm = gm0 + r;
        float val = a[r];
        if (which == 0) {
          qo[((size_t)h * M_ + gm) * D_ + gn] = f2bf(val);
        } else if (which == 1) {
          ko[((size_t)h * M_ + gm) * D_ + gn] = f2bf(val);
        } else if (which == 2) {
          int bb = gm >> 10, nn = gm & (N_ - 1);     // v stored transposed (H,B,D,N)
          vTo[(((size_t)h * B_ + bb) * D_ + gn) * N_ + nn] = f2bf(val);
        } else {
          gateo[((size_t)h * M_ + gm) * D_ + gn] =
              f2bf(1.0f / (1.0f + __expf(-val)));
        }
      }
    }
  }
}

// -------- flash attention per (h,b): WG = 128 query rows, key blocks of 32 --------
__global__ __launch_bounds__(256) void flash_attn_kernel(
    const bf16* __restrict__ q, const bf16* __restrict__ k,
    const bf16* __restrict__ vT, const int* __restrict__ mask,
    float* __restrict__ O) {
  __shared__ __align__(16) bf16 KtL[32 * 264];   // K block, row-major [m][e]
  __shared__ __align__(16) bf16 VtL[256 * 40];   // V^T block [e][m]
  __shared__ __align__(16) bf16 PsL[8 * 16 * 40];// per-wave P staging
  __shared__ float kmL[32];

  const int tid = threadIdx.x;
  const int wid = tid >> 5;
  const int lane = tid & 31, nloc = lane & 15, hf = lane >> 4;
  const int hb = blockIdx.y;                 // h*8 + b
  const int b = hb & 7;
  const int qb = blockIdx.x;
  const size_t hbofs = (size_t)hb * N_ * D_;
  const bf16* qbase = q + hbofs;
  const bf16* kbase = k + hbofs;
  const bf16* vbase = vT + hbofs;
  const int m0 = qb * 128 + wid * 16;        // this wave's 16 query rows

  v16bf qa[8];
#pragma unroll
  for (int kk = 0; kk < 8; ++kk) qa[kk] = load_frag_a(qbase, D_, m0, kk * 32);

  float qm[8];
#pragma unroll
  for (int r = 0; r < 8; ++r) qm[r] = (float)mask[b * N_ + m0 + hf * 8 + r];

  v8f zero = {};
  v8f o[16];
#pragma unroll
  for (int t = 0; t < 16; ++t) o[t] = zero;
  float l[8], mx[8];
#pragma unroll
  for (int r = 0; r < 8; ++r) { l[r] = 0.0f; mx[r] = -1e30f; }

  const int krow = tid >> 3;           // 0..31
  const int kcol = (tid & 7) * 32;     // stride-32 chunks of D

  for (int kb = 0; kb < N_ / 32; ++kb) {
    __syncthreads();                   // previous iteration's compute done
    {
      // K tile (32 x 256) and V^T tile (256 x 32): async global->LDS
      const bf16* g = kbase + (size_t)(kb * 32 + krow) * D_ + kcol;
      bf16* s = KtL + krow * 264 + kcol;
      async_ld_b128(s, g, 0);
      async_ld_b128(s, g, 16);
      async_ld_b128(s, g, 32);
      async_ld_b128(s, g, 48);
      const bf16* gv = vbase + (size_t)tid * N_ + kb * 32;
      bf16* sv = VtL + tid * 40;
      async_ld_b128(sv, gv, 0);
      async_ld_b128(sv, gv, 16);
      async_ld_b128(sv, gv, 32);
      async_ld_b128(sv, gv, 48);
      if (tid < 32) kmL[tid] = (float)mask[b * N_ + kb * 32 + tid];
    }
    wait_async0();
    __syncthreads();

    // S = Q K^T  (16 x 32 strip, two 16x16 n-tiles)
    v8f s0 = zero, s1 = zero;
#pragma unroll
    for (int kk = 0; kk < 8; ++kk) {
      s0 = wmma_bf16(qa[kk], load_frag_b(KtL, 264, 0, kk * 32), s0);
      s1 = wmma_bf16(qa[kk], load_frag_b(KtL, 264, 16, kk * 32), s1);
    }
    float km0 = kmL[nloc], km1 = kmL[16 + nloc];

    v8f rm;
#pragma unroll
    for (int r = 0; r < 8; ++r) {
      float a0 = s0[r] * 0.0625f + (1.0f - qm[r] * km0) * (-1e9f);
      float a1 = s1[r] * 0.0625f + (1.0f - qm[r] * km1) * (-1e9f);
      s0[r] = a0;
      s1[r] = a1;
      rm[r] = fmaxf(a0, a1);
    }
    // row max across the 16-lane group (rows of this half stay inside the group)
#pragma unroll
    for (int ofs = 8; ofs >= 1; ofs >>= 1)
#pragma unroll
      for (int r = 0; r < 8; ++r) rm[r] = fmaxf(rm[r], __shfl_xor(rm[r], ofs));

    float corr[8];
    v8f p0, p1;
#pragma unroll
    for (int r = 0; r < 8; ++r) {
      float mn = fmaxf(mx[r], rm[r]);
      corr[r] = __expf(mx[r] - mn);
      mx[r] = mn;
      p0[r] = __expf(s0[r] - mn) * (qm[r] * km0);   // multiplicative mask, exact zeros
      p1[r] = __expf(s1[r] - mn) * (qm[r] * km1);
    }
    v8f rs;
#pragma unroll
    for (int r = 0; r < 8; ++r) rs[r] = p0[r] + p1[r];
#pragma unroll
    for (int ofs = 8; ofs >= 1; ofs >>= 1)
#pragma unroll
      for (int r = 0; r < 8; ++r) rs[r] += __shfl_xor(rs[r], ofs);

    v8f cv;
#pragma unroll
    for (int r = 0; r < 8; ++r) {
      l[r] = l[r] * corr[r] + rs[r];
      cv[r] = corr[r];
    }
#pragma unroll
    for (int t = 0; t < 16; ++t) o[t] *= cv;

    // stage P to per-wave LDS in A-source layout, then O += P V
    bf16* ps = PsL + wid * (16 * 40);
#pragma unroll
    for (int r = 0; r < 8; ++r) {
      ps[(hf * 8 + r) * 40 + nloc] = f2bf(p0[r]);
      ps[(hf * 8 + r) * 40 + 16 + nloc] = f2bf(p1[r]);
    }
    v16bf pa = load_frag_a(ps, 40, 0, 0);
#pragma unroll
    for (int t = 0; t < 16; ++t)
      o[t] = wmma_bf16(pa, load_frag_b(VtL, 40, t * 16, 0), o[t]);
  }

  float invl[8];
#pragma unroll
  for (int r = 0; r < 8; ++r) invl[r] = (l[r] > 0.0f) ? (1.0f / l[r]) : 0.0f;
  float* obase = O + hbofs;
#pragma unroll
  for (int t = 0; t < 16; ++t)
#pragma unroll
    for (int r = 0; r < 8; ++r)
      obase[(size_t)(m0 + hf * 8 + r) * D_ + t * 16 + nloc] = o[t][r] * invl[r];
}

// -------- gate * O + x, per-head LN, concat to (B,N,D*H) bf16 --------
__global__ __launch_bounds__(256) void headfuse_kernel(
    const float* __restrict__ O, const bf16* __restrict__ gate,
    const float* __restrict__ x, const float* __restrict__ lnr_g,
    const float* __restrict__ lnr_b, bf16* __restrict__ concat) {
  __shared__ float red[256];
  int row = blockIdx.x, t = threadIdx.x;
  float xv = x[(size_t)row * D_ + t];
  float gw = lnr_g[t], gb = lnr_b[t];
  for (int h = 0; h < H_; ++h) {
    size_t idx = ((size_t)h * M_ + row) * D_ + t;
    float val = O[idx] * bf2f(gate[idx]) + xv;
    float mu = block_reduce_sum(val, red) * (1.0f / D_);
    float d = val - mu;
    float var = block_reduce_sum(d * d, red) * (1.0f / D_);
    float y = d * rsqrtf(var + EPS_) * gw + gb;
    concat[(size_t)row * (D_ * H_) + t * H_ + h] = f2bf(y);
  }
}

// -------- final projection: concat(8192x1024) @ out_w + bias + x -> tmp fp32 --------
__global__ __launch_bounds__(256) void final_gemm_kernel(
    const bf16* __restrict__ concat, const bf16* __restrict__ owT,
    const float* __restrict__ out_b, const float* __restrict__ x,
    float* __restrict__ tmp) {
  __shared__ __align__(16) bf16 As[2 * 128 * LDT];
  __shared__ __align__(16) bf16 Bs[2 * 128 * LDT];
  const int m_blk = blockIdx.y * 128;
  const int n_blk = blockIdx.x * 128;
  v8f acc[8];
  v8f zero = {};
#pragma unroll
  for (int i = 0; i < 8; ++i) acc[i] = zero;
  gemm_mainloop(concat, D_ * H_, owT, D_ * H_, D_ * H_, m_blk, n_blk, As, Bs, acc);

  const int tid = threadIdx.x;
  const int wid = tid >> 5, wm = wid >> 1, wn = wid & 1;
  const int lane = tid & 31, nloc = lane & 15, hf = lane >> 4;
#pragma unroll
  for (int i = 0; i < 2; ++i) {
#pragma unroll
    for (int j = 0; j < 4; ++j) {
      v8f a = acc[i * 4 + j];
      int gm0 = m_blk + wm * 32 + i * 16 + hf * 8;
      int gn = n_blk + wn * 64 + j * 16 + nloc;
#pragma unroll
      for (int r = 0; r < 8; ++r) {
        size_t off = (size_t)(gm0 + r) * D_ + gn;
        tmp[off] = a[r] + out_b[gn] + x[off];
      }
    }
  }
}

__global__ __launch_bounds__(256) void ln_out_kernel(
    const float* __restrict__ tmp, const float* __restrict__ g,
    const float* __restrict__ bb, const int* __restrict__ mask,
    float* __restrict__ out) {
  __shared__ float red[256];
  int row = blockIdx.x, t = threadIdx.x;
  float v = tmp[(size_t)row * D_ + t];
  float mu = block_reduce_sum(v, red) * (1.0f / D_);
  float d = v - mu;
  float var = block_reduce_sum(d * d, red) * (1.0f / D_);
  float y = d * rsqrtf(var + EPS_) * g[t] + bb[t];
  out[(size_t)row * D_ + t] = y * (float)mask[row];
}

extern "C" void kernel_launch(void* const* d_in, const int* in_sizes, int n_in,
                              void* d_out, int out_size, void* d_ws, size_t ws_size,
                              hipStream_t stream) {
  (void)in_sizes; (void)n_in; (void)out_size; (void)ws_size;
  const float* x     = (const float*)d_in[0];
  const int*   mask  = (const int*)d_in[1];
  const float* wq    = (const float*)d_in[2];
  const float* wk    = (const float*)d_in[3];
  const float* wv    = (const float*)d_in[4];
  const float* wg    = (const float*)d_in[5];
  const float* out_w = (const float*)d_in[6];
  const float* out_b = (const float*)d_in[7];
  const float* ln_g  = (const float*)d_in[8];
  const float* ln_b  = (const float*)d_in[9];
  const float* lnr_g = (const float*)d_in[10];
  const float* lnr_b = (const float*)d_in[11];
  const float* lno_g = (const float*)d_in[12];
  const float* lno_b = (const float*)d_in[13];
  float* out = (float*)d_out;

  // workspace carve-up (~127 MiB total)
  char* w = (char*)d_ws;
  auto carve = [&](size_t bytes) {
    char* p = w;
    w += (bytes + 255) & ~(size_t)255;
    return p;
  };
  bf16*  xn     = (bf16*)carve((size_t)M_ * D_ * 2);
  bf16*  wqT    = (bf16*)carve((size_t)H_ * D_ * D_ * 2);
  bf16*  wkT    = (bf16*)carve((size_t)H_ * D_ * D_ * 2);
  bf16*  wvT    = (bf16*)carve((size_t)H_ * D_ * D_ * 2);
  bf16*  wgT    = (bf16*)carve((size_t)H_ * D_ * D_ * 2);
  bf16*  owT    = (bf16*)carve((size_t)D_ * (D_ * H_) * 2);
  bf16*  qbuf   = (bf16*)carve((size_t)H_ * M_ * D_ * 2);
  bf16*  kbuf   = (bf16*)carve((size_t)H_ * M_ * D_ * 2);
  bf16*  vTbuf  = (bf16*)carve((size_t)H_ * M_ * D_ * 2);
  bf16*  gate   = (bf16*)carve((size_t)H_ * M_ * D_ * 2);
  float* Obuf   = (float*)carve((size_t)H_ * M_ * D_ * 4);
  bf16*  concat = (bf16*)carve((size_t)M_ * D_ * H_ * 2);
  float* tmp    = (float*)carve((size_t)M_ * D_ * 4);

  transpose_w_kernel<<<1024, 256, 0, stream>>>(wq, wqT);
  transpose_w_kernel<<<1024, 256, 0, stream>>>(wk, wkT);
  transpose_w_kernel<<<1024, 256, 0, stream>>>(wv, wvT);
  transpose_w_kernel<<<1024, 256, 0, stream>>>(wg, wgT);
  transpose_ow_kernel<<<1024, 256, 0, stream>>>(out_w, owT);

  ln_x_kernel<<<M_, 256, 0, stream>>>(x, ln_g, ln_b, xn);

  dim3 gp(2, 64, H_ * 4);
  proj_gemm_kernel<<<gp, 256, 0, stream>>>(xn, wqT, wkT, wvT, wgT,
                                           qbuf, kbuf, vTbuf, gate);

  dim3 ga(N_ / 128, H_ * B_);
  flash_attn_kernel<<<ga, 256, 0, stream>>>(qbuf, kbuf, vTbuf, mask, Obuf);

  headfuse_kernel<<<M_, 256, 0, stream>>>(Obuf, gate, x, lnr_g, lnr_b, concat);

  dim3 gf(2, 64);
  final_gemm_kernel<<<gf, 256, 0, stream>>>(concat, owT, out_b, x, tmp);

  ln_out_kernel<<<M_, 256, 0, stream>>>(tmp, lno_g, lno_b, mask, out);
}
